// M_EfficientSelfAtten_68375879352559
// MI455X (gfx1250) — matrix operand order repaired and checked by reading, not compiled, not run on hardware
//
#include <hip/hip_runtime.h>
#include <hip/hip_bf16.h>
#include <math.h>
#include <stdint.h>

#define B_   8
#define N_   5880
#define M1_  735
#define M2_  1764
#define M1P  736
#define M2P  1792
#define SCALE_ 0.17677669529663687f

#if defined(__has_builtin)
#if __has_builtin(__builtin_amdgcn_global_load_async_to_lds_b128) && \
    __has_builtin(__builtin_amdgcn_s_wait_asynccnt)
#define HAVE_ASYNC_LDS 1
#endif
#endif

typedef __attribute__((ext_vector_type(16))) _Float16 v16h;
typedef __attribute__((ext_vector_type(8)))  _Float16 v8h;
typedef __attribute__((ext_vector_type(8)))  float    v8f;

static __device__ __forceinline__ v8h ld8(const _Float16* p) { return *(const v8h*)p; }

static __device__ __forceinline__ v16h cat16(v8h a, v8h b) {
  v16h r;
  #pragma unroll
  for (int i = 0; i < 8; ++i) { r[i] = a[i]; r[i + 8] = b[i]; }
  return r;
}

static __device__ __forceinline__ v8f wmma16(v16h a, v16h b, v8f c) {
  return __builtin_amdgcn_wmma_f32_16x16x32_f16(false, a, false, b, (short)0, c, false, false);
}

#ifdef HAVE_ASYNC_LDS
typedef __attribute__((ext_vector_type(4))) int v4i;
typedef __attribute__((address_space(1))) v4i* gv4i_p;   // global int4*
typedef __attribute__((address_space(3))) v4i* lv4i_p;   // LDS int4*

// copy one contiguous 2KB K-block (32 keys x 32 d, f16) global -> LDS, async
static __device__ __forceinline__ void async_copy_2k(const _Float16* __restrict__ src,
                                                     _Float16* __restrict__ dst, int lane) {
  #pragma unroll
  for (int t = 0; t < 4; ++t) {
    const _Float16* s = src + (size_t)(t * 32 + lane) * 8;   // 16B per lane per issue
    _Float16* d = dst + (size_t)(t * 32 + lane) * 8;
    __builtin_amdgcn_global_load_async_to_lds_b128(
        (gv4i_p)(uintptr_t)s,
        (lv4i_p)(uint32_t)(uintptr_t)d,
        0, 0);
  }
}
#endif

// ---------------- weight convert + transpose (64x64 fp32 -> f16, Wt[n][k]) --------
__global__ void cvt_w(const float* __restrict__ w0, const float* __restrict__ w1,
                      const float* __restrict__ w2, const float* __restrict__ w3,
                      _Float16* __restrict__ dst) {
  const float* src = (blockIdx.x == 0) ? w0 : (blockIdx.x == 1) ? w1
                   : (blockIdx.x == 2) ? w2 : w3;
  _Float16* d = dst + (size_t)blockIdx.x * 4096;
  for (int i = threadIdx.x; i < 4096; i += blockDim.x) {
    int k = i >> 6, n = i & 63;
    d[n * 64 + k] = (_Float16)src[k * 64 + n];
  }
}

// ---------------- conv weight repack: OIHW fp32 -> [o][ (ky*kk+kx)*Cin+ci ] f16 ----
__global__ void cvt_conv_w(const float* __restrict__ w, _Float16* __restrict__ wt,
                           int O, int Cin, int kk) {
  int K = Cin * kk * kk;
  int i = blockIdx.x * blockDim.x + threadIdx.x;
  if (i >= O * K) return;
  int o = i / K, k = i % K;
  int ci = k % Cin; int p = k / Cin;
  int kx = p % kk;  int ky = p / kk;
  wt[(size_t)o * K + k] = (_Float16)w[(((size_t)o * Cin + ci) * kk + ky) * kk + kx];
}

// ---------------- fp32 -> f16 elementwise ----------------
__global__ void cvt_h(const float* __restrict__ in, _Float16* __restrict__ out, int n) {
  int i = blockIdx.x * blockDim.x + threadIdx.x;
  if (i < n) out[i] = (_Float16)in[i];
}

// ---------------- im2col WMMA GEMM conv (stride == kernel, channel-last) -----------
__global__ void conv_gemm_k(const _Float16* __restrict__ xh, const _Float16* __restrict__ wt,
                            const float* __restrict__ bias, float* __restrict__ out,
                            int tok_off, int Win, int Cin, int kk, int stride,
                            int Ho, int Wo, int t_off, int out_rows) {
  int lane = threadIdx.x & 31, lo = lane & 15, hi = lane >> 4;
  int ncol = blockIdx.y * 64 + (threadIdx.x >> 5) * 16 + lo;   // output channel
  int K = Cin * kk * kk;
  int HW = Ho * Wo;
  int Mtot = B_ * HW;
  int mb = blockIdx.x * 16;
  int mrow = min(mb + lo, Mtot - 1);
  int b = mrow / HW, s = mrow % HW;
  int y0 = (s / Wo) * stride, x0 = (s % Wo) * stride;
  const _Float16* abase = xh + ((size_t)b * N_ + tok_off) * 64 + ((size_t)y0 * Win + x0) * Cin;
  int seglen = kk * Cin;          // contiguous halfs per ky segment (mult of 256)
  int segstride = Win * Cin;      // memory stride between ky segments
  const _Float16* brow = wt + (size_t)ncol * K;
  float bv = bias[ncol];
  v8f c;
  #pragma unroll
  for (int r = 0; r < 8; ++r) c[r] = bv;
  for (int ks = 0; ks < K; ks += 32) {
    int ka = ks + hi * 8;
    int kb = ks + 16 + hi * 8;
    int kya = ka / seglen, ra = ka % seglen;
    int kyb = kb / seglen, rb = kb % seglen;
    v16h a = cat16(ld8(abase + (size_t)kya * segstride + ra),
                   ld8(abase + (size_t)kyb * segstride + rb));
    v16h bm = cat16(ld8(brow + ks + hi * 8), ld8(brow + ks + 16 + hi * 8));
    c = wmma16(a, bm, c);
  }
  int grp = Cin >> 6;
  #pragma unroll
  for (int r = 0; r < 8; ++r) {
    int m = mb + r + 8 * hi;
    if (m < Mtot) {
      int bb = m / HW, ss = m % HW;
      int cch = ncol / grp;
      int tt = (ncol % grp) * HW + ss;
      out[((size_t)bb * out_rows + t_off + tt) * 64 + cch] = c[r];
    }
  }
}

// ---------------- copy passthrough tokens (tem3) into both branches ----------------
__global__ void tail_copy(const float* __restrict__ x, float* __restrict__ x1raw,
                          float* __restrict__ x2raw) {
  int idx = blockIdx.x * blockDim.x + threadIdx.x;
  if (idx >= B_ * 392 * 64) return;
  int c = idx & 63; int t = (idx >> 6) % 392; int b = idx / (392 * 64);
  float v = x[((size_t)b * N_ + 5488 + t) * 64 + c];
  x1raw[((size_t)b * M1_ + 343 + t) * 64 + c] = v;
  x2raw[((size_t)b * M2_ + 1372 + t) * 64 + c] = v;
}

// ---------------- LayerNorm(64) + exact GELU, one wave per row, f16 out ------------
__global__ void ln_gelu(const float* __restrict__ in, _Float16* __restrict__ out,
                        int rows, const float* __restrict__ gamma,
                        const float* __restrict__ beta) {
  int lane = threadIdx.x & 31;
  int row = blockIdx.x * (blockDim.x >> 5) + (threadIdx.x >> 5);
  if (row >= rows) return;
  const float* p = in + (size_t)row * 64;
  float e0 = p[lane], e1 = p[lane + 32];
  float s = e0 + e1;
  #pragma unroll
  for (int m = 1; m < 32; m <<= 1) s += __shfl_xor(s, m);
  float mean = s * (1.0f / 64.0f);
  float d0 = e0 - mean, d1 = e1 - mean;
  float v = d0 * d0 + d1 * d1;
  #pragma unroll
  for (int m = 1; m < 32; m <<= 1) v += __shfl_xor(v, m);
  float rs = rsqrtf(v * (1.0f / 64.0f) + 1e-5f);
  float y0 = d0 * rs * gamma[lane] + beta[lane];
  float y1 = d1 * rs * gamma[lane + 32] + beta[lane + 32];
  y0 = 0.5f * y0 * (1.0f + erff(y0 * 0.70710678118654752f));
  y1 = 0.5f * y1 * (1.0f + erff(y1 * 0.70710678118654752f));
  out[(size_t)row * 64 + lane] = (_Float16)y0;
  out[(size_t)row * 64 + lane + 32] = (_Float16)y1;
}

// ---------------- 16x16 tile of  A(Mx64,f16) @ Wt(64x64,f16) ----------------------
static __device__ __forceinline__ v8f gemm_row64(const _Float16* __restrict__ A, int mclamp,
                                                 const _Float16* __restrict__ Bt, int ncol,
                                                 float biasv, int lo, int hi) {
  const _Float16* ar = A + (size_t)mclamp * 64;
  const _Float16* br = Bt + (size_t)ncol * 64;
  v16h a0 = cat16(ld8(ar + hi * 8),      ld8(ar + 16 + hi * 8));
  v16h a1 = cat16(ld8(ar + 32 + hi * 8), ld8(ar + 48 + hi * 8));
  v16h b0 = cat16(ld8(br + hi * 8),      ld8(br + 16 + hi * 8));
  v16h b1 = cat16(ld8(br + 32 + hi * 8), ld8(br + 48 + hi * 8));
  v8f c;
  #pragma unroll
  for (int r = 0; r < 8; ++r) c[r] = biasv;
  c = wmma16(a0, b0, c);
  c = wmma16(a1, b1, c);
  return c;
}

__global__ void gemm_q_k(const _Float16* __restrict__ xh, const _Float16* __restrict__ wt,
                         const float* __restrict__ bias, _Float16* __restrict__ qh) {
  int lane = threadIdx.x & 31, lo = lane & 15, hi = lane >> 4;
  int nb = (threadIdx.x >> 5) * 16;
  int mb = blockIdx.x * 16;
  v8f c = gemm_row64(xh, min(mb + lo, B_ * N_ - 1), wt, nb + lo, bias[nb + lo], lo, hi);
  #pragma unroll
  for (int r = 0; r < 8; ++r) {
    int m = mb + r + 8 * hi;
    qh[(size_t)m * 64 + nb + lo] = (_Float16)c[r];
  }
}

__global__ void gemm_kv_k(const _Float16* __restrict__ xh, const _Float16* __restrict__ wt,
                          const float* __restrict__ bias, _Float16* __restrict__ kh,
                          _Float16* __restrict__ vt, int Mx, int Mpad) {
  int lane = threadIdx.x & 31, lo = lane & 15, hi = lane >> 4;
  int nb = (threadIdx.x >> 5) * 16;
  int mb = blockIdx.x * 16;
  int Mtot = B_ * Mx;
  v8f c = gemm_row64(xh, min(mb + lo, Mtot - 1), wt, nb + lo, bias[nb + lo], lo, hi);
  int col = nb + lo;
  #pragma unroll
  for (int r = 0; r < 8; ++r) {
    int m = mb + r + 8 * hi;
    if (m < Mtot) {
      int b = m / Mx, t = m % Mx;
      _Float16 hv = (_Float16)c[r];
      if (col < 32) kh[((size_t)b * Mpad + t) * 32 + col] = hv;
      else          vt[((size_t)b * 32 + (col - 32)) * Mpad + t] = hv;
    }
  }
}

__global__ void gemm_proj_k(const _Float16* __restrict__ oh, const _Float16* __restrict__ wt,
                            const float* __restrict__ bias, float* __restrict__ out) {
  int lane = threadIdx.x & 31, lo = lane & 15, hi = lane >> 4;
  int nb = (threadIdx.x >> 5) * 16;
  int mb = blockIdx.x * 16;
  v8f c = gemm_row64(oh, min(mb + lo, B_ * N_ - 1), wt, nb + lo, bias[nb + lo], lo, hi);
  #pragma unroll
  for (int r = 0; r < 8; ++r) {
    int m = mb + r + 8 * hi;
    out[(size_t)m * 64 + nb + lo] = c[r];
  }
}

// ---------------- flash attention: 1 wave = 16 query rows, 32-key blocks ----------
__global__ __launch_bounds__(32) void attn_k(const _Float16* __restrict__ qh,
                                             const _Float16* __restrict__ kh,
                                             const _Float16* __restrict__ vt,
                                             _Float16* __restrict__ oh,
                                             int M, int Mpad, int head) {
  __shared__ _Float16 P[16 * 32];
#ifdef HAVE_ASYNC_LDS
  __shared__ _Float16 Kb[2][32 * 32];       // double-buffered K block via async DMA
#endif
  int lane = threadIdx.x, lo = lane & 15, hi = lane >> 4;
  int b = blockIdx.y, tile = blockIdx.x;
  int mq = tile * 16 + lo;
  const _Float16* qrow = qh + ((size_t)b * N_ + min(mq, N_ - 1)) * 64 + head * 32;
  v16h qa = cat16(ld8(qrow + hi * 16), ld8(qrow + hi * 16 + 8));

  float mi[8], li[8];
  v8f o0, o1;
  #pragma unroll
  for (int r = 0; r < 8; ++r) { mi[r] = -__builtin_inff(); li[r] = 0.f; o0[r] = 0.f; o1[r] = 0.f; }

  int nblk = (M + 31) >> 5;
  const _Float16* kbase = kh + (size_t)b * Mpad * 32;
#ifdef HAVE_ASYNC_LDS
  async_copy_2k(kbase, Kb[0], lane);
#endif
  for (int jb = 0; jb < nblk; ++jb) {
    int k0 = jb << 5;
#ifdef HAVE_ASYNC_LDS
    if (jb + 1 < nblk) {
      async_copy_2k(kbase + (size_t)(jb + 1) * 1024, Kb[(jb + 1) & 1], lane);
      __builtin_amdgcn_s_wait_asynccnt(4);   // previous block's 4 issues complete
    } else {
      __builtin_amdgcn_s_wait_asynccnt(0);
    }
    __syncthreads();
    const _Float16* kr0 = &Kb[jb & 1][lo * 32 + hi * 16];
    const _Float16* kr1 = &Kb[jb & 1][(16 + lo) * 32 + hi * 16];
#else
    const _Float16* kr0 = kh + ((size_t)b * Mpad + k0 + lo) * 32 + hi * 16;
    const _Float16* kr1 = kh + ((size_t)b * Mpad + k0 + 16 + lo) * 32 + hi * 16;
#endif
    v16h bk0 = cat16(ld8(kr0), ld8(kr0 + 8));
    v16h bk1 = cat16(ld8(kr1), ld8(kr1 + 8));
    v8f z;
    #pragma unroll
    for (int r = 0; r < 8; ++r) z[r] = 0.f;
    v8f s0 = wmma16(qa, bk0, z);
    v8f s1 = wmma16(qa, bk1, z);

    bool ok0 = (k0 + lo) < M, ok1 = (k0 + 16 + lo) < M;
    float p0[8], p1[8];
    #pragma unroll
    for (int r = 0; r < 8; ++r) {
      float a0 = ok0 ? s0[r] * SCALE_ : -__builtin_inff();
      float a1 = ok1 ? s1[r] * SCALE_ : -__builtin_inff();
      float t = fmaxf(a0, a1);
      t = fmaxf(t, __shfl_xor(t, 1));
      t = fmaxf(t, __shfl_xor(t, 2));
      t = fmaxf(t, __shfl_xor(t, 4));
      t = fmaxf(t, __shfl_xor(t, 8));
      float mnew = fmaxf(mi[r], t);
      p0[r] = __expf(a0 - mnew);
      p1[r] = __expf(a1 - mnew);
      float ps = p0[r] + p1[r];
      ps += __shfl_xor(ps, 1);
      ps += __shfl_xor(ps, 2);
      ps += __shfl_xor(ps, 4);
      ps += __shfl_xor(ps, 8);
      float alpha = __expf(mi[r] - mnew);
      li[r] = li[r] * alpha + ps;
      mi[r] = mnew;
      o0[r] *= alpha;
      o1[r] *= alpha;
    }
    #pragma unroll
    for (int r = 0; r < 8; ++r) {
      int row = r + 8 * hi;
      P[row * 32 + lo] = (_Float16)p0[r];
      P[row * 32 + 16 + lo] = (_Float16)p1[r];
    }
    __syncthreads();
    const _Float16* pr = &P[lo * 32];
    v16h pa = cat16(ld8(pr + hi * 16), ld8(pr + hi * 16 + 8));
    __syncthreads();
    const _Float16* vr0 = vt + ((size_t)b * 32 + lo) * Mpad + k0 + hi * 16;
    const _Float16* vr1 = vt + ((size_t)b * 32 + 16 + lo) * Mpad + k0 + hi * 16;
    if (jb + 1 < nblk) {                      // hint next V block toward the caches
      __builtin_prefetch(vr0 + 32, 0, 1);
      __builtin_prefetch(vr1 + 32, 0, 1);
    }
    v16h bv0 = cat16(ld8(vr0), ld8(vr0 + 8));
    v16h bv1 = cat16(ld8(vr1), ld8(vr1 + 8));
    o0 = wmma16(pa, bv0, o0);
    o1 = wmma16(pa, bv1, o1);
  }
  #pragma unroll
  for (int r = 0; r < 8; ++r) {
    int row = tile * 16 + r + 8 * hi;
    if (row < N_) {
      float inv = 1.0f / li[r];
      size_t base = ((size_t)b * N_ + row) * 64 + head * 32;
      oh[base + lo]      = (_Float16)(o0[r] * inv);
      oh[base + 16 + lo] = (_Float16)(o1[r] * inv);
    }
  }
}

extern "C" void kernel_launch(void* const* d_in, const int* in_sizes, int n_in,
                              void* d_out, int out_size, void* d_ws, size_t ws_size,
                              hipStream_t stream) {
  (void)in_sizes; (void)n_in; (void)out_size; (void)ws_size;
  const float* x      = (const float*)d_in[0];
  const float* q_w    = (const float*)d_in[1];
  const float* q_b    = (const float*)d_in[2];
  const float* kv1_w  = (const float*)d_in[3];
  const float* kv1_b  = (const float*)d_in[4];
  const float* kv2_w  = (const float*)d_in[5];
  const float* kv2_b  = (const float*)d_in[6];
  const float* proj_w = (const float*)d_in[7];
  const float* proj_b = (const float*)d_in[8];
  const float* n1_g   = (const float*)d_in[21];
  const float* n1_b   = (const float*)d_in[22];
  const float* n2_g   = (const float*)d_in[23];
  const float* n2_b   = (const float*)d_in[24];

  char* ws = (char*)d_ws;
  size_t off = 0;
  auto alloc = [&](size_t bytes) -> void* {
    void* p = ws + off;
    off = (off + bytes + 255) & ~(size_t)255;
    return p;
  };
  _Float16* xh   = (_Float16*)alloc((size_t)B_ * N_ * 64 * 2);
  _Float16* qh   = (_Float16*)alloc((size_t)B_ * N_ * 64 * 2);
  _Float16* ohb  = (_Float16*)alloc((size_t)B_ * N_ * 64 * 2);
  float*    x1raw= (float*)   alloc((size_t)B_ * M1_ * 64 * 4);
  float*    x2raw= (float*)   alloc((size_t)B_ * M2_ * 64 * 4);
  _Float16* x1h  = (_Float16*)alloc((size_t)B_ * M1_ * 64 * 2);
  _Float16* x2h  = (_Float16*)alloc((size_t)B_ * M2_ * 64 * 2);
  _Float16* wt   = (_Float16*)alloc((size_t)4 * 4096 * 2);
  _Float16* kh1  = (_Float16*)alloc((size_t)B_ * M1P * 32 * 2);
  _Float16* vt1  = (_Float16*)alloc((size_t)B_ * 32 * M1P * 2);
  _Float16* kh2  = (_Float16*)alloc((size_t)B_ * M2P * 32 * 2);
  _Float16* vt2  = (_Float16*)alloc((size_t)B_ * 32 * M2P * 2);
  _Float16* cw1  = (_Float16*)alloc((size_t)64  * 4096 * 2);
  _Float16* cw2  = (_Float16*)alloc((size_t)64  * 1024 * 2);
  _Float16* cw3  = (_Float16*)alloc((size_t)128 * 2048 * 2);
  _Float16* cw4  = (_Float16*)alloc((size_t)128 * 512  * 2);
  _Float16* cw5  = (_Float16*)alloc((size_t)256 * 1024 * 2);
  _Float16* cw6  = (_Float16*)alloc((size_t)256 * 256  * 2);

  cvt_w<<<4, 256, 0, stream>>>(q_w, kv1_w, kv2_w, proj_w, wt);
  _Float16* q_wt    = wt;
  _Float16* kv1_wt  = wt + 4096;
  _Float16* kv2_wt  = wt + 8192;
  _Float16* proj_wt = wt + 12288;

  int nx = B_ * N_ * 64;
  cvt_h<<<(nx + 255) / 256, 256, 0, stream>>>(x, xh, nx);

  cvt_conv_w<<<(64*4096  + 255)/256, 256, 0, stream>>>((const float*)d_in[9],  cw1,  64,  64, 8);
  cvt_conv_w<<<(64*1024  + 255)/256, 256, 0, stream>>>((const float*)d_in[11], cw2,  64,  64, 4);
  cvt_conv_w<<<(128*2048 + 255)/256, 256, 0, stream>>>((const float*)d_in[13], cw3, 128, 128, 4);
  cvt_conv_w<<<(128*512  + 255)/256, 256, 0, stream>>>((const float*)d_in[15], cw4, 128, 128, 2);
  cvt_conv_w<<<(256*1024 + 255)/256, 256, 0, stream>>>((const float*)d_in[17], cw5, 256, 256, 2);
  cvt_conv_w<<<(256*256  + 255)/256, 256, 0, stream>>>((const float*)d_in[19], cw6, 256, 256, 1);

  // im2col WMMA GEMM convs -> pre-LN token buffers
  conv_gemm_k<<<dim3(25, 1), 128, 0, stream>>>(xh, cw1, (const float*)d_in[10], x1raw,    0, 56,  64, 8, 8,  7,  7,   0, M1_);
  conv_gemm_k<<<dim3(25, 2), 128, 0, stream>>>(xh, cw3, (const float*)d_in[14], x1raw, 3136, 28, 128, 4, 4,  7,  7,  49, M1_);
  conv_gemm_k<<<dim3(25, 4), 128, 0, stream>>>(xh, cw5, (const float*)d_in[18], x1raw, 4704, 14, 256, 2, 2,  7,  7, 147, M1_);
  conv_gemm_k<<<dim3(98, 1), 128, 0, stream>>>(xh, cw2, (const float*)d_in[12], x2raw,    0, 56,  64, 4, 4, 14, 14,   0, M2_);
  conv_gemm_k<<<dim3(98, 2), 128, 0, stream>>>(xh, cw4, (const float*)d_in[16], x2raw, 3136, 28, 128, 2, 2, 14, 14, 196, M2_);
  conv_gemm_k<<<dim3(98, 4), 128, 0, stream>>>(xh, cw6, (const float*)d_in[20], x2raw, 4704, 14, 256, 1, 1, 14, 14, 588, M2_);
  tail_copy<<<(8*392*64 + 255)/256, 256, 0, stream>>>(x, x1raw, x2raw);

  ln_gelu<<<(B_*M1_ + 7)/8, 256, 0, stream>>>(x1raw, x1h, B_*M1_, n1_g, n1_b);
  ln_gelu<<<(B_*M2_ + 7)/8, 256, 0, stream>>>(x2raw, x2h, B_*M2_, n2_g, n2_b);

  gemm_q_k <<<(B_*N_) / 16, 128, 0, stream>>>(xh, q_wt, q_b, qh);
  gemm_kv_k<<<(B_*M1_ + 15)/16, 128, 0, stream>>>(x1h, kv1_wt, kv1_b, kh1, vt1, M1_, M1P);
  gemm_kv_k<<<(B_*M2_ + 15)/16, 128, 0, stream>>>(x2h, kv2_wt, kv2_b, kh2, vt2, M2_, M2P);

  dim3 ag((N_ + 15) / 16, B_);
  attn_k<<<ag, 32, 0, stream>>>(qh, kh1, vt1, ohb, M1_, M1P, 0);
  attn_k<<<ag, 32, 0, stream>>>(qh, kh2, vt2, ohb, M2_, M2P, 1);

  gemm_proj_k<<<(B_*N_) / 16, 128, 0, stream>>>(ohb, proj_wt, proj_b, (float*)d_out);
}
